// GPNConv_81080392614287
// MI455X (gfx1250) — matrix operand order repaired and checked by reading, not compiled, not run on hardware
//
#include <hip/hip_runtime.h>

// ---------------------------------------------------------------------------
// GPNConv for MI455X (gfx1250, wave32, WMMA)
//   h   = x + segment_sum(x[col], row)        (copy + hw f32 atomics)
//   out = h @ W^T + b                         (bf16 WMMA, f32 accum)
// ---------------------------------------------------------------------------

#define N_NODES 100000
#define N_EDGES 600000
#define D_IN    128
#define D_OUT   512

typedef __attribute__((ext_vector_type(16))) __bf16 v16bf;
typedef __attribute__((ext_vector_type(8)))  float  v8f;

union FragBF16 {
    uint4 q[2];
    v16bf v;
};

// round-to-nearest-even f32 -> bf16, packed pairwise into a dword
static __device__ __forceinline__ unsigned pack2_bf16(float a, float b) {
    union { float f; unsigned u; } ua, ub;
    ua.f = a; ub.f = b;
    unsigned ra = (ua.u + 0x7FFFu + ((ua.u >> 16) & 1u)) >> 16;
    unsigned rb = (ub.u + 0x7FFFu + ((ub.u >> 16) & 1u)) >> 16;
    return (ra & 0xFFFFu) | (rb << 16);
}

// ---------------------------------------------------------------------------
// Kernel 1: h = x   (12.8M floats as float4, fully coalesced)
// grid = 12500 x 256 threads, 1 float4 per thread
// ---------------------------------------------------------------------------
__global__ __launch_bounds__(256)
void residual_copy_kernel(const float* __restrict__ x, float* __restrict__ h) {
    int i = blockIdx.x * blockDim.x + threadIdx.x;          // float4 index
    const float4* src = reinterpret_cast<const float4*>(x);
    float4* dst = reinterpret_cast<float4*>(h);
    dst[i] = src[i];
}

// ---------------------------------------------------------------------------
// Kernel 2: scatter-add  h[row[e]] += x[col[e]]
// 32 lanes per edge; each lane owns 4 contiguous columns (float4 gather,
// 4x hardware global_atomic_add_f32).
// grid = 75000 x 256 threads  (600000 edges * 32 lanes)
// ---------------------------------------------------------------------------
__global__ __launch_bounds__(256)
void edge_scatter_kernel(const float* __restrict__ x,
                         const int* __restrict__ edge_index,
                         float* __restrict__ h) {
    int g    = blockIdx.x * blockDim.x + threadIdx.x;
    int e    = g >> 5;          // edge id
    int lane = g & 31;          // column chunk
    int row = edge_index[e];             // destination node
    int col = edge_index[N_EDGES + e];   // source node

    const float4 v = *reinterpret_cast<const float4*>(x + (size_t)col * D_IN + lane * 4);
    float* dst = h + (size_t)row * D_IN + lane * 4;
    unsafeAtomicAdd(dst + 0, v.x);
    unsafeAtomicAdd(dst + 1, v.y);
    unsafeAtomicAdd(dst + 2, v.z);
    unsafeAtomicAdd(dst + 3, v.w);
}

// ---------------------------------------------------------------------------
// Kernel 3: out = h @ W^T + b   via v_wmma_f32_16x16x32_bf16
//
// Block: 256 threads = 8 wave32.  Block tile: M=32, N=128, K=128.
//   wave_m = wave_id>>2 (2 tiles of 16 rows), wave_n = wave_id&3 (32 cols,
//   i.e. 2 accumulators of 16x16).  K unrolled: 4 steps of 32.
// LDS: W slice [128 n][128 k] bf16 (32 KB) + h tile [32 m][128 k] bf16 (8 KB).
// Grid: (512/128, 100000/32) = (4, 3125) -- exact, no masking (EXEC all-1s).
// ---------------------------------------------------------------------------
__global__ __launch_bounds__(256)
void gemm_bf16_wmma_kernel(const float* __restrict__ h,
                           const float* __restrict__ W,
                           const float* __restrict__ b,
                           float* __restrict__ out) {
    __shared__ __attribute__((aligned(16))) unsigned short lds_w[128 * 128]; // [n][k]
    __shared__ __attribute__((aligned(16))) unsigned short lds_h[32 * 128];  // [m][k]

    const int tid     = threadIdx.x;
    const int lane    = tid & 31;
    const int wave_id = tid >> 5;
    const int wave_m  = wave_id >> 2;   // 0..1
    const int wave_n  = wave_id & 3;    // 0..3
    const int half    = lane >> 4;      // 0 or 1
    const int l16     = lane & 15;

    const int n_base = blockIdx.x * 128;   // global N offset of block
    const int m_base = blockIdx.y * 32;    // global M offset of block

    // ---- Stage W slice: 128 rows (n) x 128 k, f32 -> bf16 -----------------
    // 128*128/4 = 4096 float4 loads, 16 per thread, coalesced per n-row.
    #pragma unroll
    for (int i = 0; i < 16; ++i) {
        int idx = i * 256 + tid;           // float4 index in tile
        int nl  = idx >> 5;                // 0..127
        int k4  = idx & 31;                // 0..31 (float4 within row)
        const float4 v = *reinterpret_cast<const float4*>(
            W + (size_t)(n_base + nl) * D_IN + k4 * 4);
        uint2 p;
        p.x = pack2_bf16(v.x, v.y);
        p.y = pack2_bf16(v.z, v.w);
        *reinterpret_cast<uint2*>(&lds_w[nl * 128 + k4 * 4]) = p;
    }

    // ---- Stage h tile: 32 rows x 128 k, f32 -> bf16 -----------------------
    #pragma unroll
    for (int i = 0; i < 4; ++i) {
        int idx = i * 256 + tid;           // float4 index in tile (0..1023)
        int ml  = idx >> 5;                // 0..31
        int k4  = idx & 31;
        const float4 v = *reinterpret_cast<const float4*>(
            h + (size_t)(m_base + ml) * D_IN + k4 * 4);
        uint2 p;
        p.x = pack2_bf16(v.x, v.y);
        p.y = pack2_bf16(v.z, v.w);
        *reinterpret_cast<uint2*>(&lds_h[ml * 128 + k4 * 4]) = p;
    }

    __syncthreads();

    // ---- WMMA main loop ---------------------------------------------------
    v8f acc0 = {};
    v8f acc1 = {};

    const int m_local = wave_m * 16 + l16;        // A row for this lane

    #pragma unroll
    for (int ks = 0; ks < 4; ++ks) {              // K step of 32
        // A fragment (16x32 bf16): lanes 0-15 hold K 0-7 & 16-23 of the step,
        // lanes 16-31 hold K 8-15 & 24-31. Two 16B LDS loads.
        FragBF16 a;
        {
            const unsigned short* p = &lds_h[m_local * 128 + ks * 32 + half * 8];
            a.q[0] = *reinterpret_cast<const uint4*>(p);        // K +0..7
            a.q[1] = *reinterpret_cast<const uint4*>(p + 16);   // K +16..23
        }

        // B fragments (32x16 bf16): lane holds col n=l16, K half*8+0..7 &
        // 16+half*8+0..7 within the step (matches A). Two 16B LDS loads each.
        #pragma unroll
        for (int nt = 0; nt < 2; ++nt) {
            const int nl = wave_n * 32 + nt * 16 + l16;
            const unsigned short* p = &lds_w[nl * 128 + ks * 32 + half * 8];
            FragBF16 bfr;
            bfr.q[0] = *reinterpret_cast<const uint4*>(p);      // K +0..7
            bfr.q[1] = *reinterpret_cast<const uint4*>(p + 16); // K +16..23
            if (nt == 0) {
                acc0 = __builtin_amdgcn_wmma_f32_16x16x32_bf16(
                    false, a.v, false, bfr.v, (short)0, acc0, false, false);
            } else {
                acc1 = __builtin_amdgcn_wmma_f32_16x16x32_bf16(
                    false, a.v, false, bfr.v, (short)0, acc1, false, false);
            }
        }
    }

    // ---- Epilogue: add bias, store ---------------------------------------
    // D layout: lane holds col n = l16; VGPR r -> row m = half*8 + r.
    const int m_row0 = m_base + wave_m * 16 + half * 8;
    #pragma unroll
    for (int nt = 0; nt < 2; ++nt) {
        const int n_glob = n_base + wave_n * 32 + nt * 16 + l16;
        const float bias = b[n_glob];
        const v8f acc = (nt == 0) ? acc0 : acc1;
        #pragma unroll
        for (int r = 0; r < 8; ++r) {
            out[(size_t)(m_row0 + r) * D_OUT + n_glob] = acc[r] + bias;
        }
    }
}

// ---------------------------------------------------------------------------
// Launch
// ---------------------------------------------------------------------------
extern "C" void kernel_launch(void* const* d_in, const int* in_sizes, int n_in,
                              void* d_out, int out_size, void* d_ws, size_t ws_size,
                              hipStream_t stream) {
    const float* x          = (const float*)d_in[0];
    const int*   edge_index = (const int*)d_in[1];
    const float* W          = (const float*)d_in[2];
    const float* b          = (const float*)d_in[3];
    float*       out        = (float*)d_out;
    float*       h          = (float*)d_ws;   // [N_NODES * D_IN] f32 scratch

    // 1) h = x
    {
        int n_f4 = (N_NODES * D_IN) / 4;                 // 3,200,000
        residual_copy_kernel<<<n_f4 / 256, 256, 0, stream>>>(x, h);
    }
    // 2) h[row] += x[col]  (hardware f32 atomics)
    {
        int n_thr = N_EDGES * 32;                        // 19,200,000
        edge_scatter_kernel<<<n_thr / 256, 256, 0, stream>>>(x, edge_index, h);
    }
    // 3) out = h @ W^T + b  (bf16 WMMA)
    {
        dim3 grid(D_OUT / 128, N_NODES / 32);            // (4, 3125)
        gemm_bf16_wmma_kernel<<<grid, 256, 0, stream>>>(h, W, b, out);
    }
}
